// AttentionFusion_83751862272456
// MI455X (gfx1250) — matrix operand order repaired and checked by reading, not compiled, not run on hardware
//
#include <hip/hip_runtime.h>
#include <hip/hip_bf16.h>

// ---------------------------------------------------------------------------
// MI455X / gfx1250 implementation.
// All GEMM-shaped work (convs as implicit GEMM, reduce/k/q/v linears,
// K*Q^T, P*V) runs through v_wmma_f32_16x16x32_bf16 (wave32, f32 accum).
// A/B tiles are staged to LDS with GLOBAL_LOAD_ASYNC_TO_LDS_B128 (ASYNCcnt)
// in a double-buffered pipeline when the builtins exist; otherwise a
// synchronous global->VGPR->LDS fallback is used.
// B operand convention: N-major [Npad][Kpad] so both A and B staging are
// contiguous 16B copies (no transpose on the copy path).
// ---------------------------------------------------------------------------

typedef __attribute__((ext_vector_type(16))) __bf16 v16bf;
typedef __attribute__((ext_vector_type(8)))  __bf16 v8bf;
typedef __attribute__((ext_vector_type(8)))  float  v8f;
typedef int b128_t __attribute__((__vector_size__(16)));   // matches builtin param

#define AS1 __attribute__((address_space(1)))
#define AS3 __attribute__((address_space(3)))

#if __has_builtin(__builtin_amdgcn_global_load_async_to_lds_b128) && \
    __has_builtin(__builtin_amdgcn_s_wait_asynccnt)
#define HAS_ASYNC_LDS 1
#else
#define HAS_ASYNC_LDS 0
#endif

__device__ __forceinline__ void cp_async16(void* lds, const void* g) {
#if HAS_ASYNC_LDS
  __builtin_amdgcn_global_load_async_to_lds_b128(
      (AS1 b128_t*)(unsigned long long)g,
      (AS3 b128_t*)(unsigned long long)lds, 0, 0);
#else
  *(v8bf*)lds = *(const v8bf*)g;
#endif
}
__device__ __forceinline__ void async_wait() {
#if HAS_ASYNC_LDS
  __builtin_amdgcn_s_wait_asynccnt(0);
#endif
}

__device__ __forceinline__ v16bf cat8(v8bf lo, v8bf hi) {
  return __builtin_shufflevector(lo, hi, 0,1,2,3,4,5,6,7,8,9,10,11,12,13,14,15);
}
#define WMMA_BF16(Af, Bf, Cf) \
  __builtin_amdgcn_wmma_f32_16x16x32_bf16(false, (Af), false, (Bf), (short)0, (Cf), false, false)

// ---------------------------------------------------------------------------
// Generic bf16 WMMA GEMM.  Block = 256 threads = 8 waves (4m x 2n).
// Block tile 128x64, K-step 32, wave tile 32x32 (2x2 fragments, 4 WMMA/step).
//   A:  [M][lda]      bf16 (row-major, lda*2 % 16 == 0)
//   Bm: [Npad][ldb]   bf16 (N-major; ldb >= K, ldb*2 % 16 == 0)
// Epilogue: +bias[n], *colscale[(m>>10)*csld+n], optional f32 C store,
// optional bf16 store (cbmode 0: [m][ldcb]; 1: [m>>10][n][m&1023]).
// ---------------------------------------------------------------------------
__global__ __launch_bounds__(256)
void gemm_bf16_wmma(const __bf16* __restrict__ A, int lda, long long sA,
                    const __bf16* __restrict__ Bm, int ldb, long long sB,
                    const float* __restrict__ bias,
                    const float* __restrict__ colscale, int csld,
                    float* __restrict__ C, int ldc, long long sC,
                    __bf16* __restrict__ Cb, int ldcb, long long sCb, int cbmode,
                    int M, int N, int Npad, int K)
{
  __shared__ __bf16 As[2][128][40];   // [buf][m][k], 80B rows
  __shared__ __bf16 Bs[2][64][40];    // [buf][n][k]

  const int z = blockIdx.z;
  A  += (long long)z * sA;
  Bm += (long long)z * sB;
  if (C) C += (long long)z * sC;

  const int tid  = threadIdx.x;
  const int lane = tid & 31;
  const int wave = tid >> 5;
  const int wm   = wave & 3;          // 32-row slice
  const int wn   = wave >> 2;         // 32-col slice
  const int bm   = blockIdx.y * 128;
  const int bn   = blockIdx.x * 64;

  v8f acc00 = {}, acc01 = {}, acc10 = {}, acc11 = {};

  // staging: A 128x32 -> 16 elems/thread, B 64x32 -> 8 elems/thread
  const int arow = tid >> 1;              // 0..127
  const int akc  = (tid & 1) << 4;        // 0 or 16
  const int brn  = tid >> 2;              // 0..63 (n row)
  const int bkc  = (tid & 3) << 3;        // 0,8,16,24
  const bool aok = (bm + arow) < M;
  const bool bok = (bn + brn) < Npad;
  const __bf16* aptr = A  + (long long)(bm + arow) * lda + akc;
  const __bf16* bptr = Bm + (long long)(bn + brn) * ldb + bkc;

  auto stage = [&](int k0, int buf) {
    __bf16* al = &As[buf][arow][akc];
    if (aok) {
      cp_async16(al,     aptr + k0);
      cp_async16(al + 8, aptr + k0 + 8);
    } else {
      v8bf zz = {};
      *(v8bf*)al = zz; *(v8bf*)(al + 8) = zz;
    }
    __bf16* bl = &Bs[buf][brn][bkc];
    if (bok) cp_async16(bl, bptr + k0);
    else     { v8bf zz = {}; *(v8bf*)bl = zz; }
  };

  stage(0, 0);
  int buf = 0;
  for (int k0 = 0; k0 < K; k0 += 32) {
    async_wait();
    __syncthreads();                      // tile `buf` resident, prior reads done
    if (k0 + 32 < K) stage(k0 + 32, buf ^ 1);   // overlap next tile with compute

    const int l15 = lane & 15;
    const int kbA = (lane >> 4) << 4;     // 0/16: A lane holds k kbA..kbA+15
    const int mr0 = wm * 32 + l15, mr1 = mr0 + 16;
    v16bf a0 = cat8(*(const v8bf*)&As[buf][mr0][kbA], *(const v8bf*)&As[buf][mr0][kbA + 8]);
    v16bf a1 = cat8(*(const v8bf*)&As[buf][mr1][kbA], *(const v8bf*)&As[buf][mr1][kbA + 8]);
    const int kbB = (lane >> 4) << 4;     // 0/16: B lane holds k kbB..kbB+15
    const int n0 = wn * 32 + l15, n1 = n0 + 16;
    v16bf b0 = cat8(*(const v8bf*)&Bs[buf][n0][kbB], *(const v8bf*)&Bs[buf][n0][kbB + 8]);
    v16bf b1 = cat8(*(const v8bf*)&Bs[buf][n1][kbB], *(const v8bf*)&Bs[buf][n1][kbB + 8]);

    acc00 = WMMA_BF16(a0, b0, acc00);
    acc01 = WMMA_BF16(a0, b1, acc01);
    acc10 = WMMA_BF16(a1, b0, acc10);
    acc11 = WMMA_BF16(a1, b1, acc11);
    buf ^= 1;
  }

  // epilogue
  #pragma unroll
  for (int mi = 0; mi < 2; ++mi) {
    #pragma unroll
    for (int r = 0; r < 8; ++r) {
      int m = bm + wm * 32 + mi * 16 + r + ((lane >> 4) << 3);
      if (m >= M) continue;
      #pragma unroll
      for (int ni = 0; ni < 2; ++ni) {
        int n = bn + wn * 32 + ni * 16 + (lane & 15);
        if (n >= N) continue;
        float v;
        if (mi == 0) v = ni ? acc01[r] : acc00[r];
        else         v = ni ? acc11[r] : acc10[r];
        if (bias)     v += bias[n];
        if (colscale) v *= colscale[(m >> 10) * csld + n];
        if (C) C[(long long)m * ldc + n] = v;
        if (Cb) {
          if (cbmode == 0)
            Cb[(long long)m * ldcb + n] = (__bf16)v;
          else
            Cb[(long long)(m >> 10) * sCb + (long long)n * ldcb + (m & 1023)] = (__bf16)v;
        }
      }
    }
  }
}

// ---------------------------------------------------------------------------
// 3x3 conv (pad 1) as implicit GEMM, same WMMA core.
//   X:  NHWC bf16 [32][32][32][Cpad], Cpad % 16 == 0 (zero padded)
//   Wm: [Cout][K=9*Cpad] bf16 (N-major), Y: [32768][Cout] f32 NHWC, + bias.
// ---------------------------------------------------------------------------
__global__ __launch_bounds__(256)
void conv3x3_wmma(const __bf16* __restrict__ X, int Cpad,
                  const __bf16* __restrict__ Wm,
                  const float* __restrict__ bias,
                  float* __restrict__ Y, int Cout, int K)
{
  __shared__ __bf16 As[2][128][40];
  __shared__ __bf16 Bs[2][64][40];

  const int tid  = threadIdx.x;
  const int lane = tid & 31;
  const int wave = tid >> 5;
  const int wm   = wave & 3;
  const int wn   = wave >> 2;
  const int bm   = blockIdx.y * 128;
  const int bn   = blockIdx.x * 64;

  v8f acc00 = {}, acc01 = {}, acc10 = {}, acc11 = {};

  const int arow = tid >> 1;
  const int akc  = (tid & 1) << 4;        // 16-elem chunk, never crosses a tap
  const int brn  = tid >> 2;
  const int bkc  = (tid & 3) << 3;

  const int m    = bm + arow;             // output pixel
  const int b    = m >> 10;
  const int h    = (m >> 5) & 31;
  const int wc   = m & 31;
  const bool bok = (bn + brn) < Cout;
  const __bf16* bptr = Wm + (long long)(bn + brn) * K + bkc;

  auto stage = [&](int k0, int buf) {
    // A: im2col chunk of 16 channels of one tap
    {
      int kk  = k0 + akc;
      int tap = kk / Cpad;
      int cin = kk - tap * Cpad;
      int hh  = h + tap / 3 - 1;
      int ww  = wc + tap % 3 - 1;
      __bf16* al = &As[buf][arow][akc];
      if ((unsigned)hh < 32u && (unsigned)ww < 32u) {
        const __bf16* g = X + ((long long)((b << 5) + hh) * 32 + ww) * Cpad + cin;
        cp_async16(al,     g);
        cp_async16(al + 8, g + 8);
      } else {
        v8bf zz = {};
        *(v8bf*)al = zz; *(v8bf*)(al + 8) = zz;
      }
    }
    // B: weights (N-major)
    {
      __bf16* bl = &Bs[buf][brn][bkc];
      if (bok) cp_async16(bl, bptr + k0);
      else     { v8bf zz = {}; *(v8bf*)bl = zz; }
    }
  };

  stage(0, 0);
  int buf = 0;
  for (int k0 = 0; k0 < K; k0 += 32) {
    async_wait();
    __syncthreads();
    if (k0 + 32 < K) stage(k0 + 32, buf ^ 1);

    const int l15 = lane & 15;
    const int kbA = (lane >> 4) << 4;
    const int mr0 = wm * 32 + l15, mr1 = mr0 + 16;
    v16bf a0 = cat8(*(const v8bf*)&As[buf][mr0][kbA], *(const v8bf*)&As[buf][mr0][kbA + 8]);
    v16bf a1 = cat8(*(const v8bf*)&As[buf][mr1][kbA], *(const v8bf*)&As[buf][mr1][kbA + 8]);
    const int kbB = (lane >> 4) << 4;
    const int n0 = wn * 32 + l15, n1 = n0 + 16;
    v16bf b0 = cat8(*(const v8bf*)&Bs[buf][n0][kbB], *(const v8bf*)&Bs[buf][n0][kbB + 8]);
    v16bf b1 = cat8(*(const v8bf*)&Bs[buf][n1][kbB], *(const v8bf*)&Bs[buf][n1][kbB + 8]);

    acc00 = WMMA_BF16(a0, b0, acc00);
    acc01 = WMMA_BF16(a0, b1, acc01);
    acc10 = WMMA_BF16(a1, b0, acc10);
    acc11 = WMMA_BF16(a1, b1, acc11);
    buf ^= 1;
  }

  #pragma unroll
  for (int mi = 0; mi < 2; ++mi) {
    #pragma unroll
    for (int r = 0; r < 8; ++r) {
      int mo = bm + wm * 32 + mi * 16 + r + ((lane >> 4) << 3);
      #pragma unroll
      for (int ni = 0; ni < 2; ++ni) {
        int n = bn + wn * 32 + ni * 16 + (lane & 15);
        if (n >= Cout) continue;
        float v;
        if (mi == 0) v = ni ? acc01[r] : acc00[r];
        else         v = ni ? acc11[r] : acc10[r];
        Y[(long long)mo * Cout + n] = v + bias[n];
      }
    }
  }
}

// ---------------------------------------------------------------------------
// Glue kernels
// ---------------------------------------------------------------------------

// concat(video, spatial) NCHW -> NHWC bf16, channels padded 264 -> 288
__global__ void pack_x0(const float* __restrict__ video,
                        const float* __restrict__ spatial,
                        __bf16* __restrict__ x0, int n)
{
  int i = blockIdx.x * blockDim.x + threadIdx.x;
  if (i >= n) return;
  int c = i % 288;
  int bp = i / 288;
  int b = bp >> 10, p = bp & 1023;
  float v = 0.f;
  if (c < 256)      v = video[((long long)b * 256 + c) * 1024 + p];
  else if (c < 264) v = spatial[((long long)b * 8 + (c - 256)) * 1024 + p];
  x0[i] = (__bf16)v;
}

// OIHW conv weight -> [O][tap*Cpad + cin] bf16 (N-major), zero pad cin >= Cin
__global__ void pack_conv_w(const float* __restrict__ w, int O, int Cin, int Cpad,
                            int K, __bf16* __restrict__ Wm, int n)
{
  int i = blockIdx.x * blockDim.x + threadIdx.x;
  if (i >= n) return;
  int kk = i % K;
  int o  = i / K;
  int tap = kk / Cpad;
  int cin = kk - tap * Cpad;
  float v = 0.f;
  if (cin < Cin) v = w[((long long)o * Cin + cin) * 9 + tap];
  Wm[i] = (__bf16)v;
}

// row-major [N][K] weight -> [Npad][Kpad] bf16 (N-major), zero pad
__global__ void pack_lin_w(const float* __restrict__ w, int N, int K,
                           int Kpad, __bf16* __restrict__ Bm, int n)
{
  int i = blockIdx.x * blockDim.x + threadIdx.x;
  if (i >= n) return;
  int kk = i % Kpad;
  int nn = i / Kpad;
  float v = 0.f;
  if (nn < N && kk < K) v = w[(long long)nn * K + kk];
  Bm[i] = (__bf16)v;
}

__global__ void fill_zero16(uint4* __restrict__ p, long long n16)
{
  long long i = (long long)blockIdx.x * blockDim.x + threadIdx.x;
  if (i < n16) p[i] = make_uint4(0u, 0u, 0u, 0u);
}

// per-(b,g) mean / rstd over NHWC f32, 32 groups
__global__ void gn_stats(const float* __restrict__ Y, int C, int gc,
                         float* __restrict__ mean, float* __restrict__ rstd)
{
  int bg = blockIdx.x;
  int b = bg >> 5, g = bg & 31;
  const float* yb = Y + (long long)b * 1024 * C + g * gc;
  float s = 0.f, ss = 0.f;
  for (int p = threadIdx.x; p < 1024; p += 256) {
    const float* yp = yb + (long long)p * C;
    for (int c = 0; c < gc; ++c) { float v = yp[c]; s += v; ss += v * v; }
  }
  __shared__ float rs[256], rq[256];
  rs[threadIdx.x] = s; rq[threadIdx.x] = ss;
  __syncthreads();
  for (int off = 128; off; off >>= 1) {
    if (threadIdx.x < off) {
      rs[threadIdx.x] += rs[threadIdx.x + off];
      rq[threadIdx.x] += rq[threadIdx.x + off];
    }
    __syncthreads();
  }
  if (threadIdx.x == 0) {
    float cnt = 1024.f * (float)gc;
    float mu = rs[0] / cnt;
    float var = rq[0] / cnt - mu * mu;
    mean[bg] = mu;
    rstd[bg] = rsqrtf(var + 1e-5f);
  }
}

// groupnorm-apply + relu -> bf16 NHWC
__global__ void gn_apply_relu(const float* __restrict__ Y,
                              const float* __restrict__ mean,
                              const float* __restrict__ rstd,
                              const float* __restrict__ gam,
                              const float* __restrict__ bet,
                              __bf16* __restrict__ Aout, int C, int gc, int n)
{
  int i = blockIdx.x * blockDim.x + threadIdx.x;
  if (i >= n) return;
  int c = i % C;
  int bp = i / C;
  int b = bp >> 10;
  int g = c / gc;
  float v = (Y[i] - mean[b * 32 + g]) * rstd[b * 32 + g] * gam[c] + bet[c];
  Aout[i] = (__bf16)fmaxf(v, 0.f);
}

// gn2-apply + relu + coord channels -> vso bf16 [32][1024][288]
__global__ void build_vso(const float* __restrict__ Y2,
                          const float* __restrict__ mean,
                          const float* __restrict__ rstd,
                          const float* __restrict__ gam,
                          const float* __restrict__ bet,
                          __bf16* __restrict__ vso, int n)
{
  int i = blockIdx.x * blockDim.x + threadIdx.x;
  if (i >= n) return;
  int c = i % 288;
  int bp = i / 288;
  int b = bp >> 10, p = bp & 1023;
  float v;
  if (c < 256) {
    int g = c >> 3;
    float y = Y2[(long long)bp * 256 + c];
    v = fmaxf((y - mean[b * 32 + g]) * rstd[b * 32 + g] * gam[c] + bet[c], 0.f);
  } else if (c == 256) v = -1.f + 2.f * (float)(p & 31) / 31.f;   // xx (along w)
  else if (c == 257)   v = -1.f + 2.f * (float)(p >> 5) / 31.f;   // yy (along h)
  else                 v = 0.f;
  vso[i] = (__bf16)v;
}

// max over l of txt (b,300,20)
__global__ void txt_max_k(const float* __restrict__ txt, float* __restrict__ txtmax, int n)
{
  int i = blockIdx.x * blockDim.x + threadIdx.x;
  if (i >= n) return;
  const float* t = txt + (long long)i * 20;
  float mx = t[0];
  for (int l = 1; l < 20; ++l) mx = fmaxf(mx, t[l]);
  txtmax[i] = mx;
}

// mod[b][j] = inc_b[j] + inc_w[j,:] . txtmax[b,:]
__global__ void mod_k(const float* __restrict__ txtmax,
                      const float* __restrict__ incw,
                      const float* __restrict__ incb,
                      float* __restrict__ modb, int n)
{
  int i = blockIdx.x * blockDim.x + threadIdx.x;
  if (i >= n) return;
  int b = i / 258, j = i % 258;
  const float* tm = txtmax + b * 300;
  const float* w  = incw + (long long)j * 300;
  float s = incb[j];
  for (int d = 0; d < 300; ++d) s += w[d] * tm[d];
  modb[i] = s;
}

// text attention: softmax over l=20, weighted_txt -> out[:,258:,:,:]
__global__ void txt_attention(const float* __restrict__ vsr,
                              const float* __restrict__ txt,
                              float* __restrict__ out)
{
  int i = blockIdx.x * blockDim.x + threadIdx.x;    // 32768 pixels
  if (i >= 32768) return;
  int b = i >> 10, p = i & 1023;
  const float* vr = vsr + (long long)i * 304;
  const float* tx = txt + (long long)b * 6000;
  float a[20];
  #pragma unroll
  for (int l = 0; l < 20; ++l) a[l] = 0.f;
  for (int d = 0; d < 300; ++d) {
    float vd = vr[d];
    const float* td = tx + d * 20;
    #pragma unroll
    for (int l = 0; l < 20; ++l) a[l] += vd * td[l];
  }
  const float inv_s = 0.05773502692f;   // 1/sqrt(300)
  float mx = a[0];
  #pragma unroll
  for (int l = 1; l < 20; ++l) mx = fmaxf(mx, a[l]);
  float sum = 0.f;
  #pragma unroll
  for (int l = 0; l < 20; ++l) { a[l] = __expf((a[l] - mx) * inv_s); sum += a[l]; }
  float inv = 1.f / sum;
  #pragma unroll
  for (int l = 0; l < 20; ++l) a[l] *= inv;
  float* ob = out + (long long)b * 558 * 1024 + 258 * 1024 + p;
  for (int d = 0; d < 300; ++d) {
    const float* td = tx + d * 20;
    float o = 0.f;
    #pragma unroll
    for (int l = 0; l < 20; ++l) o += a[l] * td[l];
    ob[(long long)d * 1024] = o;
  }
}

// row softmax over 1024 (scaled by 1/sqrt(258)), f32 in -> bf16 out
__global__ __launch_bounds__(256)
void softmax_rows(const float* __restrict__ S, __bf16* __restrict__ Sb)
{
  long long row = blockIdx.x;
  const float* r = S + row * 1024;
  float x[4];
  #pragma unroll
  for (int i = 0; i < 4; ++i) x[i] = r[threadIdx.x + i * 256];
  float mx = fmaxf(fmaxf(x[0], x[1]), fmaxf(x[2], x[3]));
  __shared__ float red[256];
  red[threadIdx.x] = mx; __syncthreads();
  for (int off = 128; off; off >>= 1) {
    if (threadIdx.x < off) red[threadIdx.x] = fmaxf(red[threadIdx.x], red[threadIdx.x + off]);
    __syncthreads();
  }
  mx = red[0]; __syncthreads();
  const float inv_s = 0.06225728063f;   // 1/sqrt(258)
  float e[4], s = 0.f;
  #pragma unroll
  for (int i = 0; i < 4; ++i) { e[i] = __expf((x[i] - mx) * inv_s); s += e[i]; }
  red[threadIdx.x] = s; __syncthreads();
  for (int off = 128; off; off >>= 1) {
    if (threadIdx.x < off) red[threadIdx.x] += red[threadIdx.x + off];
    __syncthreads();
  }
  float inv = 1.f / red[0];
  __bf16* o = Sb + row * 1024;
  #pragma unroll
  for (int i = 0; i < 4; ++i) o[threadIdx.x + i * 256] = (__bf16)(e[i] * inv);
}

// wv [32*1024][272] f32 -> out[b][d][p], d < 258
__global__ void write_video(const float* __restrict__ wv, float* __restrict__ out, int n)
{
  int i = blockIdx.x * blockDim.x + threadIdx.x;
  if (i >= n) return;
  int p = i & 1023;
  int t = i >> 10;          // b*258 + d
  int b = t / 258, d = t % 258;
  out[(long long)b * 558 * 1024 + (long long)d * 1024 + p] =
      wv[((long long)b * 1024 + p) * 272 + d];
}

// ---------------------------------------------------------------------------
// Host orchestration
// ---------------------------------------------------------------------------
extern "C" void kernel_launch(void* const* d_in, const int* in_sizes, int n_in,
                              void* d_out, int out_size, void* d_ws, size_t ws_size,
                              hipStream_t stream)
{
  (void)in_sizes; (void)n_in; (void)out_size; (void)ws_size;
  const float* spatial  = (const float*)d_in[0];
  const float* video    = (const float*)d_in[1];
  const float* txt      = (const float*)d_in[2];
  const float* conv1_w  = (const float*)d_in[3];
  const float* conv1_b  = (const float*)d_in[4];
  const float* gn1_g    = (const float*)d_in[5];
  const float* gn1_b    = (const float*)d_in[6];
  const float* conv2_w  = (const float*)d_in[7];
  const float* conv2_b  = (const float*)d_in[8];
  const float* gn2_g    = (const float*)d_in[9];
  const float* gn2_b    = (const float*)d_in[10];
  const float* reduce_w = (const float*)d_in[11];
  const float* reduce_b = (const float*)d_in[12];
  const float* k_w      = (const float*)d_in[13];
  const float* k_b      = (const float*)d_in[14];
  const float* q_w      = (const float*)d_in[15];
  const float* q_b      = (const float*)d_in[16];
  const float* v_w      = (const float*)d_in[17];
  const float* v_b      = (const float*)d_in[18];
  const float* inc_w    = (const float*)d_in[19];
  const float* inc_b    = (const float*)d_in[20];
  float* out = (float*)d_out;

  // bump allocator over ws (256B aligned)
  char* wp = (char*)d_ws;
  auto alloc = [&](size_t bytes) -> void* {
    void* p = (void*)wp;
    wp += (bytes + 255) & ~(size_t)255;
    return p;
  };
  const long long M  = 32 * 1024;                       // 32768 pixels
  __bf16* x0   = (__bf16*)alloc((size_t)M * 288 * 2);   // NHWC padded input
  __bf16* Wm1  = (__bf16*)alloc((size_t)128 * 2592 * 2);
  __bf16* Wm2  = (__bf16*)alloc((size_t)256 * 1152 * 2);
  __bf16* Br   = (__bf16*)alloc((size_t)304 * 288 * 2);
  __bf16* Bk   = (__bf16*)alloc((size_t)272 * 288 * 2);
  __bf16* Bq   = (__bf16*)alloc((size_t)272 * 288 * 2);
  __bf16* Bv   = (__bf16*)alloc((size_t)272 * 288 * 2);
  float*  y1   = (float*) alloc((size_t)M * 128 * 4);
  __bf16* a1   = (__bf16*)alloc((size_t)M * 128 * 2);
  float*  y2   = (float*) alloc((size_t)M * 256 * 4);
  __bf16* vso  = (__bf16*)alloc((size_t)M * 288 * 2);
  float*  vsr  = (float*) alloc((size_t)M * 304 * 4);
  float*  m1   = (float*) alloc(1024 * 4);
  float*  rs1  = (float*) alloc(1024 * 4);
  float*  m2   = (float*) alloc(1024 * 4);
  float*  rs2  = (float*) alloc(1024 * 4);
  float*  tmax = (float*) alloc((size_t)32 * 300 * 4);
  float*  modb = (float*) alloc((size_t)32 * 258 * 4);
  __bf16* kbf  = (__bf16*)alloc((size_t)M * 288 * 2);          // [32*1024][288]
  __bf16* qbf  = (__bf16*)alloc((size_t)M * 288 * 2);          // [32*1024][288]
  __bf16* vT   = (__bf16*)alloc((size_t)32 * 272 * 1024 * 2);  // [32][272][1024]
  float*  scf  = (float*) alloc((size_t)32 * 1024 * 1024 * 4);
  __bf16* scb  = (__bf16*)alloc((size_t)32 * 1024 * 1024 * 2);
  float*  wv   = (float*) alloc((size_t)M * 272 * 4);

  const int TPB = 256;
  auto g1 = [&](long long n) { return dim3((unsigned)((n + TPB - 1) / TPB)); };

  // ---- pack inputs / weights ----
  pack_x0<<<g1(M * 288), TPB, 0, stream>>>(video, spatial, x0, (int)(M * 288));
  pack_conv_w<<<g1(128LL * 2592), TPB, 0, stream>>>(conv1_w, 128, 264, 288, 2592, Wm1, 128 * 2592);
  pack_conv_w<<<g1(256LL * 1152), TPB, 0, stream>>>(conv2_w, 256, 128, 128, 1152, Wm2, 256 * 1152);
  pack_lin_w<<<g1(304LL * 288), TPB, 0, stream>>>(reduce_w, 300, 258, 288, Br, 304 * 288);
  pack_lin_w<<<g1(272LL * 288), TPB, 0, stream>>>(k_w, 258, 258, 288, Bk, 272 * 288);
  pack_lin_w<<<g1(272LL * 288), TPB, 0, stream>>>(q_w, 258, 258, 288, Bq, 272 * 288);
  pack_lin_w<<<g1(272LL * 288), TPB, 0, stream>>>(v_w, 258, 258, 288, Bv, 272 * 288);

  // ---- conv1 -> gn1 -> relu ----
  conv3x3_wmma<<<dim3(2, 256), TPB, 0, stream>>>(x0, 288, Wm1, conv1_b, y1, 128, 2592);
  gn_stats<<<1024, TPB, 0, stream>>>(y1, 128, 4, m1, rs1);
  gn_apply_relu<<<g1(M * 128), TPB, 0, stream>>>(y1, m1, rs1, gn1_g, gn1_b, a1, 128, 4,
                                                 (int)(M * 128));

  // ---- conv2 -> gn2 -> relu + coords -> vso ----
  conv3x3_wmma<<<dim3(4, 256), TPB, 0, stream>>>(a1, 128, Wm2, conv2_b, y2, 256, 1152);
  gn_stats<<<1024, TPB, 0, stream>>>(y2, 256, 8, m2, rs2);
  build_vso<<<g1(M * 288), TPB, 0, stream>>>(y2, m2, rs2, gn2_g, gn2_b, vso, (int)(M * 288));

  // ---- vsr = vso @ reduce_w^T + reduce_b ----
  gemm_bf16_wmma<<<dim3(5, 256, 1), TPB, 0, stream>>>(
      vso, 288, 0, Br, 288, 0, reduce_b, nullptr, 0,
      vsr, 304, 0, nullptr, 0, 0, 0, (int)M, 300, 304, 288);

  // ---- text path (cheap) ----
  txt_max_k<<<g1(32LL * 300), TPB, 0, stream>>>(txt, tmax, 32 * 300);
  mod_k<<<g1(32LL * 258), TPB, 0, stream>>>(tmax, inc_w, inc_b, modb, 32 * 258);
  txt_attention<<<g1(32768), TPB, 0, stream>>>(vsr, txt, out);

  // ---- zero padded bf16 attention buffers ----
  fill_zero16<<<g1(M * 288 * 2 / 16), TPB, 0, stream>>>((uint4*)kbf, M * 288 * 2 / 16);
  fill_zero16<<<g1(M * 288 * 2 / 16), TPB, 0, stream>>>((uint4*)qbf, M * 288 * 2 / 16);
  fill_zero16<<<g1(32LL * 272 * 1024 * 2 / 16), TPB, 0, stream>>>((uint4*)vT,
                                                                  32LL * 272 * 1024 * 2 / 16);

  // ---- k, q, v linears (k,q scaled by mod; v stored transposed) ----
  gemm_bf16_wmma<<<dim3(5, 256, 1), TPB, 0, stream>>>(
      vso, 288, 0, Bk, 288, 0, k_b, modb, 258,
      nullptr, 0, 0, kbf, 288, 0, 0, (int)M, 258, 272, 288);
  gemm_bf16_wmma<<<dim3(5, 256, 1), TPB, 0, stream>>>(
      vso, 288, 0, Bq, 288, 0, q_b, modb, 258,
      nullptr, 0, 0, qbf, 288, 0, 0, (int)M, 258, 272, 288);
  gemm_bf16_wmma<<<dim3(5, 256, 1), TPB, 0, stream>>>(
      vso, 288, 0, Bv, 288, 0, v_b, nullptr, 0,
      nullptr, 0, 0, vT, 1024, 272LL * 1024, 1, (int)M, 258, 272, 288);

  // ---- score = k @ q^T (per batch), softmax, weighted = P @ v ----
  gemm_bf16_wmma<<<dim3(16, 8, 32), TPB, 0, stream>>>(
      kbf, 288, 1024LL * 288, qbf, 288, 1024LL * 288, nullptr, nullptr, 0,
      scf, 1024, 1024LL * 1024, nullptr, 0, 0, 0, 1024, 1024, 1024, 288);
  softmax_rows<<<32768, TPB, 0, stream>>>(scf, scb);
  gemm_bf16_wmma<<<dim3(5, 8, 32), TPB, 0, stream>>>(
      scb, 1024, 1024LL * 1024, vT, 1024, 272LL * 1024, nullptr, nullptr, 0,
      wv, 272, 1024LL * 272, nullptr, 0, 0, 0, 1024, 258, 272, 1024);

  // ---- write weighted_video into out[:, :258, :, :] ----
  write_video<<<g1(32LL * 258 * 1024), TPB, 0, stream>>>(wv, out, 32 * 258 * 1024);
}
